// CazzyAporboTransformerBlock_35862976921711
// MI455X (gfx1250) — matrix operand order repaired and checked by reading, not compiled, not run on hardware
//
#include <hip/hip_runtime.h>
#include <hip/hip_bf16.h>

// ---------------------------------------------------------------------------
// CDNA5 / gfx1250 implementation of the CazzyAporbo transformer block.
// All contractions run on v_wmma_f32_16x16x32_bf16 (f32 accumulate).
// Operands are staged in bf16 so fragments load directly with b128 loads
// (no f32->bf16 conversion in any hot loop).  The GEMM inner loop is
// software-pipelined: slice k+1's fragments are loaded while slice k's
// WMMAs execute.
// Shapes: B=2, T=1400, C=256, H=16, D=16, FF=1024, VOCAB=1400, SLOTS=256.
// All GEMM dims are exact multiples of the tile sizes (M=175*16, N%64==0,
// K%32==0), so the inner loops carry no bounds checks at all.
// ---------------------------------------------------------------------------

typedef __attribute__((ext_vector_type(16))) __bf16 v16bf;
typedef __attribute__((ext_vector_type(8)))  __bf16 v8bf;
typedef __attribute__((ext_vector_type(8)))  float  v8f;

#define TOKENS 2800   // B*T
#define CDIM   256
#define SEQ_T  1400
#define NEG_BIG (-3.0e38f)

static __device__ __forceinline__ v8f wmma_bf16(v16bf a, v16bf b, v8f c) {
  return __builtin_amdgcn_wmma_f32_16x16x32_bf16(false, a, false, b, (short)0, c,
                                                 false, false);
}

static __device__ __forceinline__ v16bf cat8(v8bf lo, v8bf hi) {
  return __builtin_shufflevector(lo, hi, 0, 1, 2, 3, 4, 5, 6, 7, 8, 9, 10, 11,
                                 12, 13, 14, 15);
}

static __device__ __forceinline__ v8bf zero8() {
  v8bf z;
#pragma unroll
  for (int e = 0; e < 8; ++e) z[e] = (__bf16)0.0f;
  return z;
}

// ---------------------------------------------------------------------------
// Generic GEMM:  out[M,N] = act(A[M,K] * W[N,K]^T + bias) (+ resid)
// A and W are bf16; accumulate f32.  OUT_MODE: 0=f32, 1=bf16, 2=both.
// 128 threads = 4 waves; each wave owns a 16x64 output strip.  Double-
// buffered k-slices: loads of slice k+1 are issued before the WMMAs of
// slice k, so each WMMA group only waits on loads a full iteration old.
// ---------------------------------------------------------------------------
template <int ACT, bool RESID, int OUT_MODE>
__global__ __launch_bounds__(128) void gemm_wmma_kernel(
    const __bf16* __restrict__ A, int lda,
    const __bf16* __restrict__ Bm, int ldb,
    const float* __restrict__ bias,
    const float* __restrict__ resid, int ldr,
    float* __restrict__ outf, int ldo,
    __bf16* __restrict__ outh, int ldoh,
    int M, int N, int K) {
  const int lane = threadIdx.x & 31;
  const int wave = threadIdx.x >> 5;
  const int half = lane >> 4;
  const int l16  = lane & 15;
  const int mtile = blockIdx.x * 4 + wave;
  if (mtile * 16 >= M) return;  // wave-uniform exit (padding tile only)
  const int m0 = mtile * 16;
  const int n0 = blockIdx.y * 64;

  v8f acc[4];
#pragma unroll
  for (int nt = 0; nt < 4; ++nt)
#pragma unroll
    for (int r = 0; r < 8; ++r) acc[nt][r] = 0.0f;

  // Per-lane row bases (contiguous along K for both operands).
  const __bf16* arow = A + (size_t)(m0 + l16) * lda;
  const __bf16* brow = Bm + (size_t)(n0 + l16) * ldb;

  auto loadA = [&](int k) -> v16bf {
    return cat8(*(const v8bf*)(arow + k + half * 8),
                *(const v8bf*)(arow + k + 16 + half * 8));
  };
  auto loadB = [&](int k, int nt) -> v16bf {
    const __bf16* bp = brow + (size_t)(nt * 16) * ldb + k + half * 8;
    return cat8(*(const v8bf*)(bp), *(const v8bf*)(bp + 16));
  };

  // Prologue: slice 0 in flight.
  v16bf af = loadA(0);
  v16bf bf[4];
#pragma unroll
  for (int nt = 0; nt < 4; ++nt) bf[nt] = loadB(0, nt);

#pragma unroll 2
  for (int k0 = 0; k0 < K; k0 += 32) {
    // Next slice (clamped on the last iteration -> harmless reload).
    const int kn = (k0 + 32 < K) ? (k0 + 32) : k0;
    __builtin_prefetch(arow + kn + 32, 0, 3);
    v16bf af_n = loadA(kn);
    v16bf bf_n[4];
#pragma unroll
    for (int nt = 0; nt < 4; ++nt) bf_n[nt] = loadB(kn, nt);
    // Compute on the resident fragments while the next loads fly.
#pragma unroll
    for (int nt = 0; nt < 4; ++nt)
      acc[nt] = wmma_bf16(af, bf[nt], acc[nt]);
    af = af_n;
#pragma unroll
    for (int nt = 0; nt < 4; ++nt) bf[nt] = bf_n[nt];
  }

  // Epilogue: C/D layout -> row m = m0 + half*8 + r, col n = n0 + nt*16 + l16.
#pragma unroll
  for (int nt = 0; nt < 4; ++nt) {
    const int n = n0 + nt * 16 + l16;
    const float bv = bias ? bias[n] : 0.0f;
#pragma unroll
    for (int r = 0; r < 8; ++r) {
      const int m = m0 + half * 8 + r;
      float v = acc[nt][r] + bv;
      if (ACT == 1) v = 0.5f * v * (1.0f + erff(v * 0.70710678118654752f));
      if (RESID) v += resid[(size_t)m * ldr + n];
      if (OUT_MODE == 0 || OUT_MODE == 2) outf[(size_t)m * ldo + n] = v;
      if (OUT_MODE == 1 || OUT_MODE == 2) outh[(size_t)m * ldoh + n] = (__bf16)v;
    }
  }
}

// ---------------------------------------------------------------------------
// Fused flash attention, one wave per (b, head, 16-query tile).
// q/k/v are bf16 with row stride ldq (columns head*16 + d).  The 1/sqrt(D)
// scale is applied to the f32 scores.  AFF adds the gathered
// affinity[disease_ids[b,q], k] * 0.1 bias.  All loads are issued
// unconditionally from clamped addresses; invalid lanes masked by selects.
// Per 32-key chunk: K fragments and the V fragment are loaded up front so
// the global loads overlap the softmax VALU work.
// ---------------------------------------------------------------------------
template <bool AFF>
__global__ __launch_bounds__(32) void attn_kernel(
    const __bf16* __restrict__ qp, const __bf16* __restrict__ kp,
    const __bf16* __restrict__ vp, int ldq,
    const int* __restrict__ ids, const float* __restrict__ aff,
    __bf16* __restrict__ op, int ldo, int T) {
  const int lane = threadIdx.x;
  const int half = lane >> 4;
  const int l16  = lane & 15;
  const int b    = blockIdx.z;
  const int hh   = blockIdx.y;
  const int q0   = blockIdx.x * 16;

  __shared__ __bf16 p_sh[16][32];
  __shared__ int ids_sh[16];

  const v8bf z8 = zero8();

  // Q fragment: 16x32 A-layout, d in [0,16), K>=16 zero-padded.  Rows with
  // q>=T load a clamped row; their softmax rows are independent and the
  // final store is masked, so no value masking is needed.
  const int qm  = q0 + l16;
  const int qmc = (qm < T) ? qm : (T - 1);
  v16bf qf;
  {
    const __bf16* qrow = qp + ((size_t)b * T + qmc) * ldq + hh * 16 + half * 8;
    qf = cat8(*(const v8bf*)(qrow), z8);
  }
  if (AFF) {
    // Both halves write the same value to ids_sh[l16]; benign.
    ids_sh[l16] = (qm < T) ? ids[(size_t)b * T + qmc] : 0;
  }
  __syncthreads();

  float mrun[8], lrun[8];
  v8f oacc;
#pragma unroll
  for (int r = 0; r < 8; ++r) { mrun[r] = NEG_BIG; lrun[r] = 0.0f; oacc[r] = 0.0f; }

  for (int s0 = 0; s0 < T; s0 += 32) {
    // ---- load phase: both K tiles + V fragment --------------------------
    v16bf kfrag[2];
#pragma unroll
    for (int t = 0; t < 2; ++t) {
      const int key  = s0 + t * 16 + l16;
      const int keyc = (key < T) ? key : (T - 1);
      const bool kvld = (key < T);
      const __bf16* krow = kp + ((size_t)b * T + keyc) * ldq + hh * 16;
      const v8bf klo = *(const v8bf*)(krow + half * 8);
      kfrag[t] = cat8(kvld ? klo : z8, z8);
    }
    // V as B-fragment: N = d = l16, K = key-local = half*8 + 2v (+8 for v>=4).
    v16bf vf;
    {
      const __bf16* vcol = vp + hh * 16 + l16;
#pragma unroll
      for (int v = 0; v < 8; ++v) {
        const int ka = s0 + half * 8 + 2 * v + (v >= 4 ? 8 : 0);
        const int kb2 = ka + 1;
        const int kac = (ka < T) ? ka : (T - 1);
        const int kbc = (kb2 < T) ? kb2 : (T - 1);
        const __bf16 f0 = vcol[((size_t)b * T + kac) * ldq];
        const __bf16 f1 = vcol[((size_t)b * T + kbc) * ldq];
        vf[2 * v]     = (ka < T) ? f0 : (__bf16)0.0f;
        vf[2 * v + 1] = (kb2 < T) ? f1 : (__bf16)0.0f;
      }
    }

    // ---- scores: two back-to-back WMMAs ---------------------------------
    v8f sacc[2];
#pragma unroll
    for (int t = 0; t < 2; ++t) {
#pragma unroll
      for (int r = 0; r < 8; ++r) sacc[t][r] = 0.0f;
      sacc[t] = wmma_bf16(qf, kfrag[t], sacc[t]);
    }

    float sv[2][8];
#pragma unroll
    for (int t = 0; t < 2; ++t) {
      const int kidx  = s0 + t * 16 + l16;
      const bool kv   = kidx < T;
      const int kidxc = kv ? kidx : 0;
#pragma unroll
      for (int r = 0; r < 8; ++r) {
        float s = sacc[t][r] * 0.25f;  // 1/sqrt(16)
        if (AFF) {
          const int id = ids_sh[half * 8 + r];
          const float a = aff[(size_t)id * T + kidxc];
          s += kv ? 0.1f * a : 0.0f;
        }
        sv[t][r] = kv ? s : NEG_BIG;
      }
    }

    // ---- online softmax: row (half*8+r) spans lanes l16=0..15 in a half --
    float mnew[8];
#pragma unroll
    for (int r = 0; r < 8; ++r) {
      float red = fmaxf(sv[0][r], sv[1][r]);
#pragma unroll
      for (int msk = 1; msk < 16; msk <<= 1)
        red = fmaxf(red, __shfl_xor(red, msk, 32));
      mnew[r] = fmaxf(mrun[r], red);
      const float f = __expf(mrun[r] - mnew[r]);
      mrun[r] = mnew[r];
      lrun[r] *= f;
      oacc[r] *= f;
    }
#pragma unroll
    for (int t = 0; t < 2; ++t) {
#pragma unroll
      for (int r = 0; r < 8; ++r) {
        const float p = __expf(sv[t][r] - mnew[r]);  // masked rows -> 0
        float red = p;
#pragma unroll
        for (int msk = 1; msk < 16; msk <<= 1) red += __shfl_xor(red, msk, 32);
        lrun[r] += red;
        p_sh[half * 8 + r][t * 16 + l16] = (__bf16)p;
      }
    }
    __syncthreads();

    // P as A-fragment (16 queries x 32 keys): two 8-bf16 LDS runs per lane.
    const v8bf plo = *(const v8bf*)(&p_sh[l16][half * 8]);
    const v8bf phi = *(const v8bf*)(&p_sh[l16][16 + half * 8]);
    const v16bf pf = cat8(plo, phi);

    oacc = wmma_bf16(pf, vf, oacc);
    __syncthreads();
  }

#pragma unroll
  for (int r = 0; r < 8; ++r) {
    const int m = q0 + half * 8 + r;
    if (m < T) {
      const float inv = (lrun[r] > 0.0f) ? 1.0f / lrun[r] : 0.0f;
      op[((size_t)b * T + m) * ldo + hh * 16 + l16] = (__bf16)(oacc[r] * inv);
    }
  }
}

// ---------------------------------------------------------------------------
// LayerNorm over C=256: one wave per token, 8 elems/lane, bf16 output.
// ---------------------------------------------------------------------------
__global__ __launch_bounds__(256) void ln_kernel(
    const float* __restrict__ x, const float* __restrict__ g,
    const float* __restrict__ b, __bf16* __restrict__ y, int ntok) {
  const int lane = threadIdx.x & 31;
  const int tok = blockIdx.x * 8 + (threadIdx.x >> 5);
  if (tok >= ntok) return;
  const float* row = x + (size_t)tok * CDIM + lane * 8;
  const float4 va = *(const float4*)(row);
  const float4 vb = *(const float4*)(row + 4);
  float v[8] = {va.x, va.y, va.z, va.w, vb.x, vb.y, vb.z, vb.w};
  float s = 0.0f;
#pragma unroll
  for (int i = 0; i < 8; ++i) s += v[i];
#pragma unroll
  for (int m = 1; m < 32; m <<= 1) s += __shfl_xor(s, m, 32);
  const float mean = s * (1.0f / 256.0f);
  float s2 = 0.0f;
#pragma unroll
  for (int i = 0; i < 8; ++i) { const float d = v[i] - mean; s2 += d * d; }
#pragma unroll
  for (int m = 1; m < 32; m <<= 1) s2 += __shfl_xor(s2, m, 32);
  const float inv = rsqrtf(s2 * (1.0f / 256.0f) + 1e-5f);
  const float4 ga = *(const float4*)(g + lane * 8);
  const float4 gb = *(const float4*)(g + lane * 8 + 4);
  const float4 ba = *(const float4*)(b + lane * 8);
  const float4 bb4 = *(const float4*)(b + lane * 8 + 4);
  const float gg[8] = {ga.x, ga.y, ga.z, ga.w, gb.x, gb.y, gb.z, gb.w};
  const float bbv[8] = {ba.x, ba.y, ba.z, ba.w, bb4.x, bb4.y, bb4.z, bb4.w};
  v8bf o;
#pragma unroll
  for (int i = 0; i < 8; ++i) o[i] = (__bf16)((v[i] - mean) * inv * gg[i] + bbv[i]);
  *(v8bf*)(y + (size_t)tok * CDIM + lane * 8) = o;
}

// ---------------------------------------------------------------------------
// Softmax over 256 columns (memory-bank read weights): f32 in, bf16 out.
// ---------------------------------------------------------------------------
__global__ __launch_bounds__(256) void softmax256_kernel(
    const float* __restrict__ p, __bf16* __restrict__ y, int ntok) {
  const int lane = threadIdx.x & 31;
  const int tok = blockIdx.x * 8 + (threadIdx.x >> 5);
  if (tok >= ntok) return;
  const float* row = p + (size_t)tok * CDIM + lane * 8;
  const float4 va = *(const float4*)(row);
  const float4 vb = *(const float4*)(row + 4);
  float v[8] = {va.x, va.y, va.z, va.w, vb.x, vb.y, vb.z, vb.w};
  float mx = NEG_BIG;
#pragma unroll
  for (int i = 0; i < 8; ++i) mx = fmaxf(mx, v[i]);
#pragma unroll
  for (int m = 1; m < 32; m <<= 1) mx = fmaxf(mx, __shfl_xor(mx, m, 32));
  float s = 0.0f;
#pragma unroll
  for (int i = 0; i < 8; ++i) { v[i] = __expf(v[i] - mx); s += v[i]; }
#pragma unroll
  for (int m = 1; m < 32; m <<= 1) s += __shfl_xor(s, m, 32);
  const float inv = 1.0f / s;
  v8bf o;
#pragma unroll
  for (int i = 0; i < 8; ++i) o[i] = (__bf16)(v[i] * inv);
  *(v8bf*)(y + (size_t)tok * CDIM + lane * 8) = o;
}

// ---------------------------------------------------------------------------
// One-shot f32 -> bf16 weight conversion (8 elems/thread).
// ---------------------------------------------------------------------------
__global__ __launch_bounds__(256) void cvt_bf16_kernel(
    const float* __restrict__ src, __bf16* __restrict__ dst, int n) {
  const int i = (blockIdx.x * 256 + threadIdx.x) * 8;
  if (i >= n) return;
  const float4 a = *(const float4*)(src + i);
  const float4 b = *(const float4*)(src + i + 4);
  v8bf o;
  o[0] = (__bf16)a.x; o[1] = (__bf16)a.y; o[2] = (__bf16)a.z; o[3] = (__bf16)a.w;
  o[4] = (__bf16)b.x; o[5] = (__bf16)b.y; o[6] = (__bf16)b.z; o[7] = (__bf16)b.w;
  *(v8bf*)(dst + i) = o;
}

// Transpose-convert mem_bank[S,C] (f32) -> dst[C,S] (bf16), S=C=256.
__global__ __launch_bounds__(256) void cvt_t256_kernel(
    const float* __restrict__ src, __bf16* __restrict__ dst) {
  const int n = blockIdx.x;   // output row (= column of src)
  const int k = threadIdx.x;  // output col (= row of src)
  dst[(size_t)n * 256 + k] = (__bf16)src[(size_t)k * 256 + n];
}

// ---------------------------------------------------------------------------
// Host orchestration.
// ---------------------------------------------------------------------------
extern "C" void kernel_launch(void* const* d_in, const int* in_sizes, int n_in,
                              void* d_out, int out_size, void* d_ws,
                              size_t ws_size, hipStream_t stream) {
  (void)in_sizes; (void)n_in; (void)out_size; (void)ws_size;
  const float* x          = (const float*)d_in[0];
  const int*   ids        = (const int*)d_in[1];
  const float* in_proj_w  = (const float*)d_in[2];
  const float* in_proj_b  = (const float*)d_in[3];
  const float* out_proj_w = (const float*)d_in[4];
  const float* out_proj_b = (const float*)d_in[5];
  const float* ln1_g = (const float*)d_in[6],  *ln1_b = (const float*)d_in[7];
  const float* ln2_g = (const float*)d_in[8],  *ln2_b = (const float*)d_in[9];
  const float* ln3_g = (const float*)d_in[10], *ln3_b = (const float*)d_in[11];
  const float* gq_w = (const float*)d_in[12], *gq_b = (const float*)d_in[13];
  const float* gk_w = (const float*)d_in[14], *gk_b = (const float*)d_in[15];
  const float* gv_w = (const float*)d_in[16], *gv_b = (const float*)d_in[17];
  const float* go_w = (const float*)d_in[18], *go_b = (const float*)d_in[19];
  const float* affinity = (const float*)d_in[20];
  const float* mem_bank = (const float*)d_in[21];
  const float* read_w   = (const float*)d_in[22];
  const float* read_b   = (const float*)d_in[23];
  const float* ffn_w1 = (const float*)d_in[24], *ffn_b1 = (const float*)d_in[25];
  const float* ffn_w2 = (const float*)d_in[26], *ffn_b2 = (const float*)d_in[27];
  float* out = (float*)d_out;

  // ---- workspace layout (~21 MB) ----
  float* bx   = (float*)d_ws;                        // running residual [2800,256] f32
  float* blog = bx + (size_t)TOKENS * 256;           // read logits      [2800,256] f32
  __bf16* p16 = (__bf16*)(blog + (size_t)TOKENS * 256);
  __bf16* bh16   = p16; p16 += (size_t)TOKENS * 256;   // LN / x copies  [2800,256]
  __bf16* bqkv16 = p16; p16 += (size_t)TOKENS * 768;   // qkv            [2800,768]
  __bf16* bo16   = p16; p16 += (size_t)TOKENS * 256;   // attn out / rw  [2800,256]
  __bf16* bff16  = p16; p16 += (size_t)TOKENS * 1024;  // GELU out       [2800,1024]
  __bf16* w_in   = p16; p16 += 768 * 256;
  __bf16* w_out  = p16; p16 += 256 * 256;
  __bf16* w_gq   = p16; p16 += 256 * 256;
  __bf16* w_gk   = p16; p16 += 256 * 256;
  __bf16* w_gv   = p16; p16 += 256 * 256;
  __bf16* w_go   = p16; p16 += 256 * 256;
  __bf16* w_read = p16; p16 += 256 * 256;
  __bf16* w_memT = p16; p16 += 256 * 256;
  __bf16* w_ff1  = p16; p16 += 1024 * 256;
  __bf16* w_ff2  = p16; p16 += 256 * 1024;

  const dim3 blkG(128);
  const dim3 gN256((TOKENS / 16 + 3) / 4, 256 / 64);
  const dim3 gN768((TOKENS / 16 + 3) / 4, 768 / 64);
  const dim3 gN1024((TOKENS / 16 + 3) / 4, 1024 / 64);
  const dim3 gAttn((SEQ_T + 15) / 16, 16, 2);
  const dim3 gTok((TOKENS + 7) / 8);

  // ---- 0. one-shot weight conversion to bf16 ----
  cvt_bf16_kernel<<<768 * 256 / 2048, 256, 0, stream>>>(in_proj_w, w_in, 768 * 256);
  cvt_bf16_kernel<<<256 * 256 / 2048, 256, 0, stream>>>(out_proj_w, w_out, 256 * 256);
  cvt_bf16_kernel<<<256 * 256 / 2048, 256, 0, stream>>>(gq_w, w_gq, 256 * 256);
  cvt_bf16_kernel<<<256 * 256 / 2048, 256, 0, stream>>>(gk_w, w_gk, 256 * 256);
  cvt_bf16_kernel<<<256 * 256 / 2048, 256, 0, stream>>>(gv_w, w_gv, 256 * 256);
  cvt_bf16_kernel<<<256 * 256 / 2048, 256, 0, stream>>>(go_w, w_go, 256 * 256);
  cvt_bf16_kernel<<<256 * 256 / 2048, 256, 0, stream>>>(read_w, w_read, 256 * 256);
  cvt_bf16_kernel<<<1024 * 256 / 2048, 256, 0, stream>>>(ffn_w1, w_ff1, 1024 * 256);
  cvt_bf16_kernel<<<256 * 1024 / 2048, 256, 0, stream>>>(ffn_w2, w_ff2, 256 * 1024);
  cvt_t256_kernel<<<256, 256, 0, stream>>>(mem_bank, w_memT);

  // ---- 1. MHA ----
  ln_kernel<<<gTok, 256, 0, stream>>>(x, ln1_g, ln1_b, bh16, TOKENS);
  gemm_wmma_kernel<0, false, 1><<<gN768, blkG, 0, stream>>>(
      bh16, 256, w_in, 256, in_proj_b, nullptr, 0, nullptr, 0, bqkv16, 768,
      TOKENS, 768, 256);
  attn_kernel<false><<<gAttn, 32, 0, stream>>>(
      bqkv16, bqkv16 + 256, bqkv16 + 512, 768, nullptr, nullptr, bo16, 256, SEQ_T);
  gemm_wmma_kernel<0, true, 0><<<gN256, blkG, 0, stream>>>(
      bo16, 256, w_out, 256, out_proj_b, x, 256, bx, 256, nullptr, 0,
      TOKENS, 256, 256);

  // ---- 2. Disease-graph attention ----
  ln_kernel<<<gTok, 256, 0, stream>>>(bx, ln2_g, ln2_b, bh16, TOKENS);
  gemm_wmma_kernel<0, false, 1><<<gN256, blkG, 0, stream>>>(
      bh16, 256, w_gq, 256, gq_b, nullptr, 0, nullptr, 0, bqkv16, 768,
      TOKENS, 256, 256);
  gemm_wmma_kernel<0, false, 1><<<gN256, blkG, 0, stream>>>(
      bh16, 256, w_gk, 256, gk_b, nullptr, 0, nullptr, 0, bqkv16 + 256, 768,
      TOKENS, 256, 256);
  gemm_wmma_kernel<0, false, 1><<<gN256, blkG, 0, stream>>>(
      bh16, 256, w_gv, 256, gv_b, nullptr, 0, nullptr, 0, bqkv16 + 512, 768,
      TOKENS, 256, 256);
  attn_kernel<true><<<gAttn, 32, 0, stream>>>(
      bqkv16, bqkv16 + 256, bqkv16 + 512, 768, ids, affinity, bo16, 256, SEQ_T);
  // go projection: f32 residual update + bf16 copy of new x for the read GEMM.
  gemm_wmma_kernel<0, true, 2><<<gN256, blkG, 0, stream>>>(
      bo16, 256, w_go, 256, go_b, bx, 256, bx, 256, bh16, 256,
      TOKENS, 256, 256);

  // ---- 3. Memory-bank read ----
  gemm_wmma_kernel<0, false, 0><<<gN256, blkG, 0, stream>>>(
      bh16, 256, w_read, 256, read_b, nullptr, 0, blog, 256, nullptr, 0,
      TOKENS, 256, 256);
  softmax256_kernel<<<gTok, 256, 0, stream>>>(blog, bo16, TOKENS);
  gemm_wmma_kernel<0, true, 0><<<gN256, blkG, 0, stream>>>(
      bo16, 256, w_memT, 256, nullptr, bx, 256, bx, 256, nullptr, 0,
      TOKENS, 256, 256);

  // ---- 4. FFN ----
  ln_kernel<<<gTok, 256, 0, stream>>>(bx, ln3_g, ln3_b, bh16, TOKENS);
  gemm_wmma_kernel<1, false, 1><<<gN1024, blkG, 0, stream>>>(
      bh16, 256, w_ff1, 256, ffn_b1, nullptr, 0, nullptr, 0, bff16, 1024,
      TOKENS, 1024, 256);
  gemm_wmma_kernel<0, true, 0><<<gN256, blkG, 0, stream>>>(
      bff16, 1024, w_ff2, 1024, ffn_b2, bx, 256, out, 256, nullptr, 0,
      TOKENS, 256, 1024);
}
